// RNN_16690242913226
// MI455X (gfx1250) — matrix operand order repaired and checked by reading, not compiled, not run on hardware
//
#include <hip/hip_runtime.h>
#include <hip/hip_bf16.h>

// ---------------------------------------------------------------------------
// Types for CDNA5 WMMA (wave32): D(16x16,f32) = A(16x32,f16) x B(32x16,f16) + C
// ---------------------------------------------------------------------------
typedef _Float16 h16;
typedef __attribute__((ext_vector_type(16))) _Float16 v16h;
typedef __attribute__((ext_vector_type(8)))  float    v8f;

#define DEVINL __device__ __forceinline__

// A fragment from LDS: 16-bit A 16x32 layout (ISA 7.12.2):
// lanes 0-15: M=lane, K chunks [0..7] and [16..23]; lanes 16-31: K chunks [8..15],[24..31]
DEVINL v16h load_afrag(const h16* A, int lda, int k0, int lane) {
  const int m  = lane & 15;
  const int kb = (lane >> 4) << 3;                 // 0 or 8
  const h16* p = A + m * lda + k0 + kb;
  union { uint4 u[2]; v16h v; } t;
  t.u[0] = *(const uint4*)(p);
  t.u[1] = *(const uint4*)(p + 16);
  return t.v;
}

// B fragment from global: 16-bit B 32x16 layout:
// lanes 0-15: K chunks [0..7],[16..23]; lanes 16-31: K [8..15],[24..31]; lane%16 = column N; W stored row-major [N][K]
DEVINL v16h load_bfrag(const h16* W, int ldk, int n0, int k0, int lane) {
  const int n = n0 + (lane & 15);
  const int k = k0 + ((lane >> 4) << 3);           // 0 or 8
  const h16* p = W + (size_t)n * ldk + k;
  union { uint4 u[2]; v16h v; } t;
  t.u[0] = *(const uint4*)(p);
  t.u[1] = *(const uint4*)(p + 16);
  return t.v;
}

DEVINL v8f wmma_f16(v16h a, v16h b, v8f c) {
  return __builtin_amdgcn_wmma_f32_16x16x32_f16(false, a, false, b, (short)0, c,
                                                false, false);
}

DEVINL float sigf(float x) { return 1.0f / (1.0f + __expf(-x)); }

DEVINL float wred_sum(float s) {
  for (int o = 16; o > 0; o >>= 1) s += __shfl_xor(s, o, 32);
  return s;
}
DEVINL float wred_max(float s) {
  for (int o = 16; o > 0; o >>= 1) s = fmaxf(s, __shfl_xor(s, o, 32));
  return s;
}

// ---------------------------------------------------------------------------
// Packing / input-build kernels
// ---------------------------------------------------------------------------
__global__ __launch_bounds__(256) void pack_f16_kernel(const float* __restrict__ src,
                                                       h16* __restrict__ dst, int n) {
  for (int i = blockIdx.x * 256 + threadIdx.x; i < n; i += gridDim.x * 256)
    dst[i] = (h16)src[i];
}

// dst[N][Ki+Kh] = [Wi | Wh] rows, f32 -> f16
__global__ __launch_bounds__(256) void pack_cat_kernel(const float* __restrict__ Wi,
                                                       const float* __restrict__ Wh,
                                                       h16* __restrict__ dst,
                                                       int N, int Ki, int Kh) {
  const int K = Ki + Kh;
  const int total = N * K;
  for (int i = blockIdx.x * 256 + threadIdx.x; i < total; i += gridDim.x * 256) {
    int r = i / K, c = i % K;
    float v = (c < Ki) ? Wi[(size_t)r * Ki + c] : Wh[(size_t)r * Kh + (c - Ki)];
    dst[i] = (h16)v;
  }
}

// dst[t][b][F+LD] = concat(seq[b][t][:], lang_embed[lang[b]][:]) as f16
__global__ __launch_bounds__(256) void build_x_kernel(const float* __restrict__ seq,
                                                      const float* __restrict__ emb,
                                                      const int* __restrict__ lang,
                                                      h16* __restrict__ dst,
                                                      int T, int F, int LDv) {
  const int KXv = F + LDv;
  const size_t total = (size_t)T * 512 * KXv;
  for (size_t i = (size_t)blockIdx.x * 256 + threadIdx.x; i < total;
       i += (size_t)gridDim.x * 256) {
    int k = (int)(i % KXv);
    size_t rb = i / KXv;
    int b = (int)(rb % 512);
    int t = (int)(rb / 512);
    float v = (k < F) ? seq[((size_t)b * T + t) * F + k]
                      : emb[(size_t)lang[b] * LDv + (k - F)];
    dst[i] = (h16)v;
  }
}

// ---------------------------------------------------------------------------
// Fused BiLSTM direction kernel. One block = 16 batch rows, loops all T steps.
// gates(16 x 1024) = [x_t | h] (16 x KT) @ Wcat^T  via WMMA, cell update in regs.
// Wave w owns j-tiles {w, w+8} so i/f/g/o for a hidden index share a wave.
// ---------------------------------------------------------------------------
template <int KX, int T>
__global__ __launch_bounds__(256) void lstm_kernel(
    const h16* __restrict__ X,          // [T][512][KX]
    const h16* __restrict__ Wcat,       // [1024][KX+256]
    const float* __restrict__ bi, const float* __restrict__ bh,
    h16* __restrict__ enc,              // [T][512][512], write cols [colOff,+256)
    int colOff, int reverse,
    float* __restrict__ hOut, float* __restrict__ cOut) {  // (512,256) or null
  constexpr int KT = KX + 256;
  __shared__ h16 Albs[16][KT];
  const int tid = threadIdx.x, lane = tid & 31, w = tid >> 5;
  const int b0 = blockIdx.x * 16;

  for (int i = tid; i < 16 * 256; i += 256)
    Albs[i >> 8][KX + (i & 255)] = (h16)0.0f;
  float creg[2][8];
  for (int ji = 0; ji < 2; ++ji)
    for (int r = 0; r < 8; ++r) creg[ji][r] = 0.f;
  v8f vzero = {};
  __syncthreads();

  for (int s = 0; s < T; ++s) {
    const int t = reverse ? (T - 1 - s) : s;
    // stage x_t (f16, contiguous per row) into A-buffer cols [0,KX)
    constexpr int NV = 16 * KX / 8;
    for (int i = tid; i < NV; i += 256) {
      int r = i / (KX / 8), c = i % (KX / 8);
      *(uint4*)&Albs[r][c * 8] =
          *(const uint4*)(X + ((size_t)t * 512 + b0 + r) * KX + c * 8);
    }
    __syncthreads();

    v8f acc[2][4];
    for (int ji = 0; ji < 2; ++ji)
      for (int g = 0; g < 4; ++g) acc[ji][g] = vzero;
    for (int kt = 0; kt < KT / 32; ++kt) {
      v16h af = load_afrag(&Albs[0][0], KT, kt * 32, lane);
      for (int ji = 0; ji < 2; ++ji)
        for (int g = 0; g < 4; ++g) {
          const int nt = g * 16 + (w + 8 * ji);
          v16h bf = load_bfrag(Wcat, KT, nt * 16, kt * 32, lane);
          acc[ji][g] = wmma_f16(af, bf, acc[ji][g]);
        }
    }
    __syncthreads();

    for (int ji = 0; ji < 2; ++ji) {
      const int j = (w + 8 * ji) * 16 + (lane & 15);
      const float bI = bi[j] + bh[j];
      const float bF = bi[256 + j] + bh[256 + j];
      const float bG = bi[512 + j] + bh[512 + j];
      const float bO = bi[768 + j] + bh[768 + j];
      for (int r = 0; r < 8; ++r) {
        float iv = sigf(acc[ji][0][r] + bI);
        float fv = sigf(acc[ji][1][r] + bF);
        float gv = tanhf(acc[ji][2][r] + bG);
        float ov = sigf(acc[ji][3][r] + bO);
        float c = fv * creg[ji][r] + iv * gv;
        creg[ji][r] = c;
        float h = ov * tanhf(c);
        const int m = r + 8 * (lane >> 4);
        Albs[m][KX + j] = (h16)h;
        enc[((size_t)t * 512 + b0 + m) * 512 + colOff + j] = (h16)h;
        if (hOut != nullptr && s == T - 1) {
          hOut[(size_t)(b0 + m) * 256 + j] = h;
          cOut[(size_t)(b0 + m) * 256 + j] = c;
        }
      }
    }
    __syncthreads();
  }
}

// ---------------------------------------------------------------------------
// K/V projection: out[M][512] = A[M][512] @ W^T + bias  (all f16, f32 accum)
// grid.x = M/16 row tiles, grid.y = N/128, wave = one 16x16 tile
// ---------------------------------------------------------------------------
__global__ __launch_bounds__(256) void kv_gemm_kernel(const h16* __restrict__ A,
                                                      const h16* __restrict__ W,
                                                      const float* __restrict__ bias,
                                                      h16* __restrict__ out) {
  __shared__ h16 As[16][512];
  const int tid = threadIdx.x, lane = tid & 31, w = tid >> 5;
  const size_t m0 = (size_t)blockIdx.x * 16;
  const uint4* src = (const uint4*)(A + m0 * 512);
  uint4* dst = (uint4*)&As[0][0];
  for (int i = tid; i < 1024; i += 256) dst[i] = src[i];
  __syncthreads();
  const int nt = blockIdx.y * 8 + w;
  v8f acc = {};
  for (int kt = 0; kt < 16; ++kt) {
    v16h af = load_afrag(&As[0][0], 512, kt * 32, lane);
    v16h bf = load_bfrag(W, 512, nt * 16, kt * 32, lane);
    acc = wmma_f16(af, bf, acc);
  }
  const int n = nt * 16 + (lane & 15);
  const float bb = bias[n];
  for (int r = 0; r < 8; ++r) {
    const int m = r + 8 * (lane >> 4);
    out[(m0 + m) * 512 + n] = (h16)(acc[r] + bb);
  }
}

// ---------------------------------------------------------------------------
// Fused decoder: one block = 16 batch rows, all 64 steps internal.
// ---------------------------------------------------------------------------
struct DecArgs {
  const h16 *Wcell;                     // [2048][768]
  const h16 *Wq_c, *Wq_t;               // [512][512]
  const h16 *OWc, *OWt;                 // [512][512]
  const h16 *Wout;                      // [256][1024]
  const float *cbi, *cbh;               // (2048)
  const float *bq_c, *bq_t;             // (512)
  const float *ob_c, *ob_t;             // (512)
  const float *outb;                    // (256)
  const h16 *Kc, *Vc;                   // [64][512][512]
  const h16 *Kt, *Vt;                   // [32][512][512]
  const float *chT, *ccT, *thT, *tcT;   // (512,256)
  float *out;                           // (512,64,256)
};

__global__ __launch_bounds__(256) void decoder_kernel(DecArgs a) {
  __shared__ h16 Albs[16][768];         // [x(256) | h(512)] A operand (24 KB)
  __shared__ float sw[16][96];          // attention weights char(64)+tag(32)
  __shared__ h16 bufA[16][512];         // ctx ; aliased as f32 logits[16][256]
  __shared__ h16 bufB[16][512];         // q / ca / ta
  float* lg = (float*)&bufA[0][0];

  const int tid = threadIdx.x, lane = tid & 31, w = tid >> 5;
  const int b0 = blockIdx.x * 16;
  const float qscale = 0.044194173824159216f;   // 1/sqrt(512)
  v8f vzero = {};

  // init: x = 0, h = concat(chT, thT), c regs = concat(ccT, tcT)
  for (int i = tid; i < 16 * 256; i += 256) Albs[i >> 8][i & 255] = (h16)0.0f;
  for (int i = tid; i < 16 * 512; i += 256) {
    int m = i >> 9, j = i & 511;
    float v = (j < 256) ? a.chT[(size_t)(b0 + m) * 256 + j]
                        : a.thT[(size_t)(b0 + m) * 256 + (j - 256)];
    Albs[m][256 + j] = (h16)v;
  }
  float creg[4][8];
  for (int ji = 0; ji < 4; ++ji) {
    const int j = (w + 8 * ji) * 16 + (lane & 15);
    for (int r = 0; r < 8; ++r) {
      const int m = r + 8 * (lane >> 4);
      creg[ji][r] = (j < 256) ? a.ccT[(size_t)(b0 + m) * 256 + j]
                              : a.tcT[(size_t)(b0 + m) * 256 + (j - 256)];
    }
  }
  __syncthreads();

  for (int t = 0; t < 64; ++t) {
    __syncthreads();
    v8f lacc[2];
    lacc[0] = vzero; lacc[1] = vzero;

    // ---- gates = [x|h] @ Wcell^T : 16x2048, K=768 -------------------------
    {
      v8f acc[4][4];
      for (int ji = 0; ji < 4; ++ji)
        for (int g = 0; g < 4; ++g) acc[ji][g] = vzero;
      for (int kt = 0; kt < 24; ++kt) {
        v16h af = load_afrag(&Albs[0][0], 768, kt * 32, lane);
        for (int ji = 0; ji < 4; ++ji)
          for (int g = 0; g < 4; ++g) {
            const int nt = g * 32 + (w + 8 * ji);
            v16h bf = load_bfrag(a.Wcell, 768, nt * 16, kt * 32, lane);
            acc[ji][g] = wmma_f16(af, bf, acc[ji][g]);
          }
      }
      __syncthreads();
      for (int ji = 0; ji < 4; ++ji) {
        const int j = (w + 8 * ji) * 16 + (lane & 15);
        const float bI = a.cbi[j] + a.cbh[j];
        const float bF = a.cbi[512 + j] + a.cbh[512 + j];
        const float bG = a.cbi[1024 + j] + a.cbh[1024 + j];
        const float bO = a.cbi[1536 + j] + a.cbh[1536 + j];
        for (int r = 0; r < 8; ++r) {
          float iv = sigf(acc[ji][0][r] + bI);
          float fv = sigf(acc[ji][1][r] + bF);
          float gv = tanhf(acc[ji][2][r] + bG);
          float ov = sigf(acc[ji][3][r] + bO);
          float c = fv * creg[ji][r] + iv * gv;
          creg[ji][r] = c;
          float h = ov * tanhf(c);
          const int m = r + 8 * (lane >> 4);
          Albs[m][256 + j] = (h16)h;
        }
      }
      __syncthreads();
    }

    // ---- q_c = (h @ Wq_c^T + bq)*scale -> bufB ---------------------------
    {
      v8f qa[4];
      for (int i = 0; i < 4; ++i) qa[i] = vzero;
      for (int kt = 0; kt < 16; ++kt) {
        v16h af = load_afrag(&Albs[0][256], 768, kt * 32, lane);
        for (int i = 0; i < 4; ++i) {
          v16h bf = load_bfrag(a.Wq_c, 512, (w + 8 * i) * 16, kt * 32, lane);
          qa[i] = wmma_f16(af, bf, qa[i]);
        }
      }
      for (int i = 0; i < 4; ++i) {
        const int n = (w + 8 * i) * 16 + (lane & 15);
        const float b = a.bq_c[n];
        for (int r = 0; r < 8; ++r) {
          const int m = r + 8 * (lane >> 4);
          bufB[m][n] = (h16)((qa[i][r] + b) * qscale);
        }
      }
    }
    __syncthreads();

    // ---- char scores: s[r][l] = q_c[r] . Kc[l][b0+r] ----------------------
    for (int task = w; task < 16 * 64; task += 8) {
      const int r = task & 15, l = task >> 4;
      const h16* kp = a.Kc + ((size_t)l * 512 + (b0 + r)) * 512 + lane * 16;
      const h16* qp = &bufB[r][lane * 16];
      float s = 0.f;
      for (int i = 0; i < 16; ++i) s += (float)qp[i] * (float)kp[i];
      s = wred_sum(s);
      if (lane == 0) sw[r][l] = s;
    }
    __syncthreads();

    // ---- q_t -> bufB (q_c dead) ------------------------------------------
    {
      v8f qa[4];
      for (int i = 0; i < 4; ++i) qa[i] = vzero;
      for (int kt = 0; kt < 16; ++kt) {
        v16h af = load_afrag(&Albs[0][256], 768, kt * 32, lane);
        for (int i = 0; i < 4; ++i) {
          v16h bf = load_bfrag(a.Wq_t, 512, (w + 8 * i) * 16, kt * 32, lane);
          qa[i] = wmma_f16(af, bf, qa[i]);
        }
      }
      __syncthreads();
      for (int i = 0; i < 4; ++i) {
        const int n = (w + 8 * i) * 16 + (lane & 15);
        const float b = a.bq_t[n];
        for (int r = 0; r < 8; ++r) {
          const int m = r + 8 * (lane >> 4);
          bufB[m][n] = (h16)((qa[i][r] + b) * qscale);
        }
      }
    }
    __syncthreads();

    // ---- tag scores -------------------------------------------------------
    for (int task = w; task < 16 * 32; task += 8) {
      const int r = task & 15, l = task >> 4;
      const h16* kp = a.Kt + ((size_t)l * 512 + (b0 + r)) * 512 + lane * 16;
      const h16* qp = &bufB[r][lane * 16];
      float s = 0.f;
      for (int i = 0; i < 16; ++i) s += (float)qp[i] * (float)kp[i];
      s = wred_sum(s);
      if (lane == 0) sw[r][64 + l] = s;
    }
    __syncthreads();

    // ---- softmax per row, per attention ----------------------------------
    if (tid < 32) {
      const int r = tid & 15, seg = tid >> 4;
      const int base = seg ? 64 : 0, len = seg ? 32 : 64;
      float mx = -1e30f;
      for (int l = 0; l < len; ++l) mx = fmaxf(mx, sw[r][base + l]);
      float sm = 0.f;
      for (int l = 0; l < len; ++l) {
        float e = __expf(sw[r][base + l] - mx);
        sw[r][base + l] = e;
        sm += e;
      }
      const float inv = 1.f / sm;
      for (int l = 0; l < len; ++l) sw[r][base + l] *= inv;
    }
    __syncthreads();

    // ---- ctx_c -> bufA ----------------------------------------------------
    {
      const int e = tid * 2;
      for (int r = 0; r < 16; ++r) {
        float a0 = 0.f, a1 = 0.f;
        for (int l = 0; l < 64; ++l) {
          const float wgt = sw[r][l];
          const h16* vp = a.Vc + ((size_t)l * 512 + (b0 + r)) * 512 + e;
          a0 += wgt * (float)vp[0];
          a1 += wgt * (float)vp[1];
        }
        bufA[r][e] = (h16)a0;
        bufA[r][e + 1] = (h16)a1;
      }
    }
    __syncthreads();

    // ---- ca = ctx_c @ OWc^T + ob_c -> bufB -------------------------------
    {
      v8f ca[4];
      for (int i = 0; i < 4; ++i) ca[i] = vzero;
      for (int kt = 0; kt < 16; ++kt) {
        v16h af = load_afrag(&bufA[0][0], 512, kt * 32, lane);
        for (int i = 0; i < 4; ++i) {
          v16h bf = load_bfrag(a.OWc, 512, (w + 8 * i) * 16, kt * 32, lane);
          ca[i] = wmma_f16(af, bf, ca[i]);
        }
      }
      for (int i = 0; i < 4; ++i) {
        const int n = (w + 8 * i) * 16 + (lane & 15);
        const float b = a.ob_c[n];
        for (int r = 0; r < 8; ++r) {
          const int m = r + 8 * (lane >> 4);
          bufB[m][n] = (h16)(ca[i][r] + b);
        }
      }
    }
    __syncthreads();

    // ---- logits += ca @ Wout[:, :512]^T ----------------------------------
    for (int kt = 0; kt < 16; ++kt) {
      v16h af = load_afrag(&bufB[0][0], 512, kt * 32, lane);
      for (int i = 0; i < 2; ++i) {
        v16h bf = load_bfrag(a.Wout, 1024, (w + 8 * i) * 16, kt * 32, lane);
        lacc[i] = wmma_f16(af, bf, lacc[i]);
      }
    }
    __syncthreads();

    // ---- ctx_t -> bufA ----------------------------------------------------
    {
      const int e = tid * 2;
      for (int r = 0; r < 16; ++r) {
        float a0 = 0.f, a1 = 0.f;
        for (int l = 0; l < 32; ++l) {
          const float wgt = sw[r][64 + l];
          const h16* vp = a.Vt + ((size_t)l * 512 + (b0 + r)) * 512 + e;
          a0 += wgt * (float)vp[0];
          a1 += wgt * (float)vp[1];
        }
        bufA[r][e] = (h16)a0;
        bufA[r][e + 1] = (h16)a1;
      }
    }
    __syncthreads();

    // ---- ta = ctx_t @ OWt^T + ob_t -> bufB -------------------------------
    {
      v8f ta[4];
      for (int i = 0; i < 4; ++i) ta[i] = vzero;
      for (int kt = 0; kt < 16; ++kt) {
        v16h af = load_afrag(&bufA[0][0], 512, kt * 32, lane);
        for (int i = 0; i < 4; ++i) {
          v16h bf = load_bfrag(a.OWt, 512, (w + 8 * i) * 16, kt * 32, lane);
          ta[i] = wmma_f16(af, bf, ta[i]);
        }
      }
      __syncthreads();
      for (int i = 0; i < 4; ++i) {
        const int n = (w + 8 * i) * 16 + (lane & 15);
        const float b = a.ob_t[n];
        for (int r = 0; r < 8; ++r) {
          const int m = r + 8 * (lane >> 4);
          bufB[m][n] = (h16)(ta[i][r] + b);
        }
      }
    }
    __syncthreads();

    // ---- logits += ta @ Wout[:, 512:]^T ; + out_b -> lg (f32 in bufA) -----
    for (int kt = 0; kt < 16; ++kt) {
      v16h af = load_afrag(&bufB[0][0], 512, kt * 32, lane);
      for (int i = 0; i < 2; ++i) {
        v16h bf = load_bfrag(a.Wout, 1024, (w + 8 * i) * 16, 512 + kt * 32, lane);
        lacc[i] = wmma_f16(af, bf, lacc[i]);
      }
    }
    __syncthreads();
    for (int i = 0; i < 2; ++i) {
      const int n = (w + 8 * i) * 16 + (lane & 15);
      const float b = a.outb[n];
      for (int r = 0; r < 8; ++r) {
        const int m = r + 8 * (lane >> 4);
        lg[m * 256 + n] = lacc[i][r] + b;
      }
    }
    __syncthreads();

    // ---- entmax-1.5 via bisection; write output + next x ------------------
    for (int rr = w * 2; rr < w * 2 + 2; ++rr) {
      float z[8];
      float mx = -1e30f;
      for (int i = 0; i < 8; ++i) {
        z[i] = 0.5f * lg[rr * 256 + lane * 8 + i];
        mx = fmaxf(mx, z[i]);
      }
      mx = wred_max(mx);
      for (int i = 0; i < 8; ++i) z[i] -= mx;
      float lo = -1.f, hi = 0.f;
      for (int it = 0; it < 40; ++it) {
        const float mid = 0.5f * (lo + hi);
        float s = 0.f;
        for (int i = 0; i < 8; ++i) {
          const float d = z[i] - mid;
          s += (d > 0.f) ? d * d : 0.f;
        }
        s = wred_sum(s);
        if (s >= 1.f) lo = mid; else hi = mid;
      }
      const float tau = 0.5f * (lo + hi);
      for (int i = 0; i < 8; ++i) {
        const float d = z[i] - tau;
        const float p = (d > 0.f) ? d * d : 0.f;
        a.out[((size_t)(b0 + rr) * 64 + t) * 256 + lane * 8 + i] = p;
        Albs[rr][lane * 8 + i] = (h16)p;
      }
    }
  }
}

// ---------------------------------------------------------------------------
// Host orchestration
// ---------------------------------------------------------------------------
extern "C" void kernel_launch(void* const* d_in, const int* in_sizes, int n_in,
                              void* d_out, int out_size, void* d_ws, size_t ws_size,
                              hipStream_t stream) {
  (void)in_sizes; (void)n_in; (void)out_size; (void)ws_size;
  const float* char_seq = (const float*)d_in[0];   // (512,64,256)
  const float* tagset   = (const float*)d_in[1];   // (512,32,128)
  const int*   lang     = (const int*)d_in[2];     // (512)
  const float* emb      = (const float*)d_in[3];   // (100,64)
  const float* cfWi = (const float*)d_in[4],  *cfWh = (const float*)d_in[5];
  const float* cfbi = (const float*)d_in[6],  *cfbh = (const float*)d_in[7];
  const float* cbWi = (const float*)d_in[8],  *cbWh = (const float*)d_in[9];
  const float* cbbi = (const float*)d_in[10], *cbbh = (const float*)d_in[11];
  const float* tfWi = (const float*)d_in[12], *tfWh = (const float*)d_in[13];
  const float* tfbi = (const float*)d_in[14], *tfbh = (const float*)d_in[15];
  const float* tbWi = (const float*)d_in[16], *tbWh = (const float*)d_in[17];
  const float* tbbi = (const float*)d_in[18], *tbbh = (const float*)d_in[19];
  const float* ceWi = (const float*)d_in[20], *ceWh = (const float*)d_in[21];
  const float* cebi = (const float*)d_in[22], *cebh = (const float*)d_in[23];
  const float* cain_w = (const float*)d_in[24], *cain_b = (const float*)d_in[25];
  const float* caout_w = (const float*)d_in[26], *caout_b = (const float*)d_in[27];
  const float* tain_w = (const float*)d_in[28], *tain_b = (const float*)d_in[29];
  const float* taout_w = (const float*)d_in[30], *taout_b = (const float*)d_in[31];
  const float* out_w = (const float*)d_in[32], *out_b = (const float*)d_in[33];

  char* ws = (char*)d_ws;
  size_t off = 0;
  auto take = [&](size_t bytes) -> void* {
    void* p = ws + off;
    off = (off + bytes + 255) & ~(size_t)255;
    return p;
  };

  h16* Wcf   = (h16*)take((size_t)1024 * 576 * 2);
  h16* Wcb   = (h16*)take((size_t)1024 * 576 * 2);
  h16* Wtf   = (h16*)take((size_t)1024 * 448 * 2);
  h16* Wtb   = (h16*)take((size_t)1024 * 448 * 2);
  h16* Wcell = (h16*)take((size_t)2048 * 768 * 2);
  h16* AWc   = (h16*)take((size_t)1536 * 512 * 2);   // [Wq;Wk;Wv] char
  h16* AWt   = (h16*)take((size_t)1536 * 512 * 2);
  h16* OWc   = (h16*)take((size_t)512 * 512 * 2);
  h16* OWt   = (h16*)take((size_t)512 * 512 * 2);
  h16* WOUT  = (h16*)take((size_t)256 * 1024 * 2);
  h16* Xc    = (h16*)take((size_t)64 * 512 * 320 * 2);
  h16* Xt    = (h16*)take((size_t)32 * 512 * 192 * 2);
  h16* EncC  = (h16*)take((size_t)64 * 512 * 512 * 2);
  h16* EncT  = (h16*)take((size_t)32 * 512 * 512 * 2);
  h16* Kc    = (h16*)take((size_t)64 * 512 * 512 * 2);
  h16* Vc    = (h16*)take((size_t)64 * 512 * 512 * 2);
  h16* Kt    = (h16*)take((size_t)32 * 512 * 512 * 2);
  h16* Vt    = (h16*)take((size_t)32 * 512 * 512 * 2);
  float* chT = (float*)take((size_t)512 * 256 * 4);
  float* ccT = (float*)take((size_t)512 * 256 * 4);
  float* thT = (float*)take((size_t)512 * 256 * 4);
  float* tcT = (float*)take((size_t)512 * 256 * 4);

  auto blocks = [](size_t n) { return (unsigned)((n + 255) / 256); };

  // weight packing (f32 -> f16, WMMA B-operand layout [N][K])
  pack_cat_kernel<<<blocks((size_t)1024 * 576), 256, 0, stream>>>(cfWi, cfWh, Wcf, 1024, 320, 256);
  pack_cat_kernel<<<blocks((size_t)1024 * 576), 256, 0, stream>>>(cbWi, cbWh, Wcb, 1024, 320, 256);
  pack_cat_kernel<<<blocks((size_t)1024 * 448), 256, 0, stream>>>(tfWi, tfWh, Wtf, 1024, 192, 256);
  pack_cat_kernel<<<blocks((size_t)1024 * 448), 256, 0, stream>>>(tbWi, tbWh, Wtb, 1024, 192, 256);
  pack_cat_kernel<<<blocks((size_t)2048 * 768), 256, 0, stream>>>(ceWi, ceWh, Wcell, 2048, 256, 512);
  pack_f16_kernel<<<blocks((size_t)1536 * 512), 256, 0, stream>>>(cain_w, AWc, 1536 * 512);
  pack_f16_kernel<<<blocks((size_t)1536 * 512), 256, 0, stream>>>(tain_w, AWt, 1536 * 512);
  pack_f16_kernel<<<blocks((size_t)512 * 512), 256, 0, stream>>>(caout_w, OWc, 512 * 512);
  pack_f16_kernel<<<blocks((size_t)512 * 512), 256, 0, stream>>>(taout_w, OWt, 512 * 512);
  pack_f16_kernel<<<blocks((size_t)256 * 1024), 256, 0, stream>>>(out_w, WOUT, 256 * 1024);

  // inputs: [t][b][feat+lang_embed] in f16
  build_x_kernel<<<blocks((size_t)64 * 512 * 320), 256, 0, stream>>>(char_seq, emb, lang, Xc, 64, 256, 64);
  build_x_kernel<<<blocks((size_t)32 * 512 * 192), 256, 0, stream>>>(tagset, emb, lang, Xt, 32, 128, 64);

  // BiLSTM encoders (whole scan fused per direction)
  lstm_kernel<320, 64><<<32, 256, 0, stream>>>(Xc, Wcf, cfbi, cfbh, EncC, 0,   0, chT, ccT);
  lstm_kernel<320, 64><<<32, 256, 0, stream>>>(Xc, Wcb, cbbi, cbbh, EncC, 256, 1, nullptr, nullptr);
  lstm_kernel<192, 32><<<32, 256, 0, stream>>>(Xt, Wtf, tfbi, tfbh, EncT, 0,   0, thT, tcT);
  lstm_kernel<192, 32><<<32, 256, 0, stream>>>(Xt, Wtb, tbbi, tbbh, EncT, 256, 1, nullptr, nullptr);

  // K/V precompute (K = enc @ Wk^T + bk, V = enc @ Wv^T + bv)
  kv_gemm_kernel<<<dim3(2048, 4), 256, 0, stream>>>(EncC, AWc + (size_t)512 * 512,  cain_b + 512,  Kc);
  kv_gemm_kernel<<<dim3(2048, 4), 256, 0, stream>>>(EncC, AWc + (size_t)1024 * 512, cain_b + 1024, Vc);
  kv_gemm_kernel<<<dim3(1024, 4), 256, 0, stream>>>(EncT, AWt + (size_t)512 * 512,  tain_b + 512,  Kt);
  kv_gemm_kernel<<<dim3(1024, 4), 256, 0, stream>>>(EncT, AWt + (size_t)1024 * 512, tain_b + 1024, Vt);

  // fused decoder
  DecArgs da;
  da.Wcell = Wcell;
  da.Wq_c = AWc; da.Wq_t = AWt;
  da.OWc = OWc; da.OWt = OWt; da.Wout = WOUT;
  da.cbi = cebi; da.cbh = cebh;
  da.bq_c = cain_b; da.bq_t = tain_b;
  da.ob_c = caout_b; da.ob_t = taout_b;
  da.outb = out_b;
  da.Kc = Kc; da.Vc = Vc; da.Kt = Kt; da.Vt = Vt;
  da.chT = chT; da.ccT = ccT; da.thT = thT; da.tcT = tcT;
  da.out = (float*)d_out;
  decoder_kernel<<<32, 256, 0, stream>>>(da);
}
